// MFMCGCNABSA_89283780149576
// MI455X (gfx1250) — matrix operand order repaired and checked by reading, not compiled, not run on hardware
//
#include <hip/hip_runtime.h>
#include <hip/hip_bf16.h>

typedef __bf16 bf16;
typedef __attribute__((ext_vector_type(16))) __bf16 v16bf;
typedef __attribute__((ext_vector_type(8)))  __bf16 bf16x8;
typedef __attribute__((ext_vector_type(8)))  float  v8f;

#define B_   32
#define S_   512
#define D_   768
#define H_   4
#define D2_  1536
#define DK_  384

static __device__ __forceinline__ v16bf cat8(bf16x8 lo, bf16x8 hi) {
    return __builtin_shufflevector(lo, hi, 0,1,2,3,4,5,6,7,8,9,10,11,12,13,14,15);
}

// ---------------------------------------------------------------------------
// f32 -> bf16 conversion, optionally strided output (to place x inside inp1)
// ---------------------------------------------------------------------------
__global__ __launch_bounds__(256)
void cvt_f32_bf16_strided(const float* __restrict__ in, bf16* __restrict__ out,
                          int rowlen, int outstride, long long n) {
    long long i = (long long)blockIdx.x * 256 + threadIdx.x;
    if (i >= n) return;
    long long r = i / rowlen;
    int       c = (int)(i % rowlen);
    out[r * (long long)outstride + c] = (bf16)in[i];
}

// ---------------------------------------------------------------------------
// denom = adj.sum(last) + 1, one block per row
// ---------------------------------------------------------------------------
__global__ __launch_bounds__(256)
void rowsum_plus1(const float* __restrict__ adj, float* __restrict__ denom, int S) {
    __shared__ float red[256];
    int row = blockIdx.x;
    const float* p = adj + (size_t)row * S;
    float s = 0.f;
    for (int j = threadIdx.x; j < S; j += 256) s += p[j];
    red[threadIdx.x] = s; __syncthreads();
    for (int off = 128; off > 0; off >>= 1) {
        if (threadIdx.x < off) red[threadIdx.x] += red[threadIdx.x + off];
        __syncthreads();
    }
    if (threadIdx.x == 0) denom[row] = red[0] + 1.0f;
}

// ---------------------------------------------------------------------------
// Tiled bf16 WMMA GEMM: C[M,N] = alpha * A[M,K] x B[K,N] (+bias[n])
//   - 256 threads = 8 waves; workgroup tile 64x128; wave tile 16x64 (4 WMMAs)
//   - LDS B tile stored transposed [n][k] so each fragment is 2x ds_load_b128
//   - TRANSB (compile-time): B accessed as row-major [N,K] (for q.kT)
//   - CBF16 (compile-time): output dtype
//   - batch offset: off(z) = (z/bdiv)*stride1 + (z%bdiv)*stride2 (elements)
// ---------------------------------------------------------------------------
template <int TRANSB, int CBF16>
__global__ __launch_bounds__(256)
void wmma_gemm_bf16(const bf16* __restrict__ A, long long a_s1, long long a_s2, int lda,
                    const bf16* __restrict__ Bm, long long b_s1, long long b_s2, int ldb,
                    const float* __restrict__ bias, float alpha,
                    void* __restrict__ Cv, long long c_s1, long long c_s2, int ldc,
                    int bdiv, int M, int N, int K) {
    __shared__ bf16 sA[64][40];    // [m][k], padded row = 80B
    __shared__ bf16 sBt[128][40];  // [n][k] transposed, padded row = 80B

    const int tid  = threadIdx.x;
    const int wave = tid >> 5;
    const int lane = tid & 31;
    const int half = lane >> 4;
    const int l16  = lane & 15;
    const int mo   = (wave & 3) * 16;   // wave M offset in tile
    const int no   = (wave >> 2) * 64;  // wave N offset in tile
    const int mBase = blockIdx.y * 64;
    const int nBase = blockIdx.x * 128;
    const int z     = blockIdx.z;
    const int zq = z / bdiv, zr = z % bdiv;

    const bf16* Ab = A  + (size_t)zq * a_s1 + (size_t)zr * a_s2;
    const bf16* Bb = Bm + (size_t)zq * b_s1 + (size_t)zr * b_s2;

    v8f acc[4];
    #pragma unroll
    for (int t = 0; t < 4; ++t) acc[t] = (v8f){0,0,0,0,0,0,0,0};

    // staging index precompute
    const int a_r  = tid >> 2;            // A tile: 64x32, 8 elems/thread
    const int a_c  = (tid & 3) * 8;
    const int bn_k = tid >> 3;            // B NN: k = 0..31, 16 n per thread
    const int bn_n = (tid & 7) * 16;
    const int bt_n = tid & 127;           // B NT: n = 0..127, 16 k per thread
    const int bt_k = (tid >> 7) * 16;

    for (int k0 = 0; k0 < K; k0 += 32) {
        const int kpf = (k0 + 32 < K) ? 32 : 0;   // branchless prefetch offset
        // ---- stage A tile 64x32 (16B vector per thread) ----
        {
            const bf16* src = Ab + (size_t)(mBase + a_r) * lda + (k0 + a_c);
            *(bf16x8*)&sA[a_r][a_c] = *(const bf16x8*)src;
            __builtin_prefetch(src + kpf, 0, 1);
        }
        // ---- stage B tile 32x128 into transposed LDS [n][k] ----
        if (!TRANSB) {
            const bf16* src = Bb + (size_t)(k0 + bn_k) * ldb + (nBase + bn_n);
            bf16x8 v0 = *(const bf16x8*)src;
            bf16x8 v1 = *(const bf16x8*)(src + 8);
            #pragma unroll
            for (int i = 0; i < 8; ++i) {
                sBt[bn_n + i][bn_k]     = v0[i];
                sBt[bn_n + 8 + i][bn_k] = v1[i];
            }
            __builtin_prefetch(src + (size_t)kpf * ldb, 0, 1);
        } else {
            const bf16* src = Bb + (size_t)(nBase + bt_n) * ldb + (k0 + bt_k);
            *(bf16x8*)&sBt[bt_n][bt_k]     = *(const bf16x8*)src;
            *(bf16x8*)&sBt[bt_n][bt_k + 8] = *(const bf16x8*)(src + 8);
            __builtin_prefetch(src + kpf, 0, 1);
        }
        __syncthreads();

        // ---- fragments: contiguous 2x16B LDS loads per CDNA5 layouts ----
        const bf16* arow = &sA[mo + l16][half * 8];
        v16bf af = cat8(*(const bf16x8*)arow, *(const bf16x8*)(arow + 16));
        #pragma unroll
        for (int t = 0; t < 4; ++t) {
            const bf16* brow = &sBt[no + t * 16 + l16][half * 8];
            v16bf bfv = cat8(*(const bf16x8*)brow, *(const bf16x8*)(brow + 16));
            acc[t] = __builtin_amdgcn_wmma_f32_16x16x32_bf16(
                         false, af, false, bfv, (short)0, acc[t], false, false);
        }
        __syncthreads();
    }

    // ---- epilogue: lane holds col l16 (per 16-tile), rows half*8+v ----
    const long long cOff = (long long)zq * c_s1 + (long long)zr * c_s2;
    #pragma unroll
    for (int t = 0; t < 4; ++t) {
        const int col = nBase + no + t * 16 + l16;
        const float bv = bias ? bias[col] : 0.f;
        #pragma unroll
        for (int v = 0; v < 8; ++v) {
            int row = mBase + mo + half * 8 + v;
            float val = acc[t][v] * alpha + bv;
            long long idx = cOff + (long long)row * ldc + col;
            if (CBF16) ((bf16*)Cv)[idx] = (bf16)val;
            else       ((float*)Cv)[idx] = val;
        }
    }
}

// ---------------------------------------------------------------------------
// GCN combine: v = relu(AxW/denom) + xW ; write bf16 (strided concat) or f32
// ---------------------------------------------------------------------------
__global__ __launch_bounds__(256)
void gcn_combine(const float* __restrict__ AxW, const float* __restrict__ xW,
                 const float* __restrict__ denom, int rowlen,
                 bf16* __restrict__ out_bf, int out_stride, int out_off,
                 float* __restrict__ out_f32, long long n) {
    long long i = (long long)blockIdx.x * 256 + threadIdx.x;
    if (i >= n) return;
    long long r = i / rowlen;
    int       c = (int)(i % rowlen);
    float v = AxW[i] / denom[r];
    v = v > 0.f ? v : 0.f;
    v += xW[i];
    if (out_bf) out_bf[r * (long long)out_stride + out_off + c] = (bf16)v;
    else        out_f32[i] = v;
}

// ---------------------------------------------------------------------------
// masked softmax over j, summed over heads; block = (b,i) row, 256 thr, S=512
// ---------------------------------------------------------------------------
__global__ __launch_bounds__(256)
void attn_softmax_sumheads(const float* __restrict__ scores,
                           const float* __restrict__ adj,
                           float* __restrict__ adj1raw) {
    __shared__ float red[256];
    int b = blockIdx.x / S_;
    int i = blockIdx.x % S_;
    int t = threadIdx.x;
    const float* arow = adj + ((size_t)b * S_ + i) * S_;
    float am0 = arow[t];
    float am1 = arow[t + 256];
    float bfac = (arow[0] != 0.0f) ? 1.0f : 0.0f;
    float acc0 = 0.f, acc1 = 0.f;
    for (int h = 0; h < H_; ++h) {
        const float* srow = scores + (((size_t)(b * H_ + h)) * S_ + i) * S_;
        float v0 = (am0 == 0.0f) ? -1e9f : srow[t];
        float v1 = (am1 == 0.0f) ? -1e9f : srow[t + 256];
        red[t] = fmaxf(v0, v1); __syncthreads();
        for (int off = 128; off > 0; off >>= 1) {
            if (t < off) red[t] = fmaxf(red[t], red[t + off]);
            __syncthreads();
        }
        float mx = red[0]; __syncthreads();
        float e0 = __expf(v0 - mx), e1 = __expf(v1 - mx);
        red[t] = e0 + e1; __syncthreads();
        for (int off = 128; off > 0; off >>= 1) {
            if (t < off) red[t] += red[t + off];
            __syncthreads();
        }
        float inv = 1.0f / red[0]; __syncthreads();
        acc0 += e0 * inv * bfac;
        acc1 += e1 * inv * bfac;
    }
    float* orow = adj1raw + ((size_t)b * S_ + i) * S_;
    orow[t]       = acc0;
    orow[t + 256] = acc1;
}

// ---------------------------------------------------------------------------
// per-batch kth-largest via 33-step bisection on float bit patterns (vals >= 0)
// ---------------------------------------------------------------------------
__global__ __launch_bounds__(256)
void topk_threshold(const float* __restrict__ adj1raw, const int* __restrict__ kptr,
                    float* __restrict__ thr) {
    __shared__ int red[256];
    __shared__ unsigned sh_lo, sh_hi;
    int b = blockIdx.x;
    int t = threadIdx.x;
    int k = *kptr;
    const float* p = adj1raw + (size_t)b * S_ * S_;
    if (t == 0) { sh_lo = 0u; sh_hi = 0x7F800000u; }
    __syncthreads();
    for (int it = 0; it < 33; ++it) {
        unsigned lo = sh_lo, hi = sh_hi;
        unsigned mid = lo + ((hi - lo) >> 1);
        int c = 0;
        for (int idx = t; idx < S_ * S_; idx += 256)
            c += (__float_as_uint(p[idx]) >= mid) ? 1 : 0;
        red[t] = c; __syncthreads();
        for (int off = 128; off > 0; off >>= 1) {
            if (t < off) red[t] += red[t + off];
            __syncthreads();
        }
        if (t == 0) { if (red[0] >= k) sh_lo = mid; else sh_hi = mid; }
        __syncthreads();
    }
    if (t == 0) thr[b] = __uint_as_float(sh_lo);
}

// ---------------------------------------------------------------------------
// adj1 = select(raw)*raw (binarize>=thr, symmetrize, diag=1) -> bf16 + denom1
// ---------------------------------------------------------------------------
__global__ __launch_bounds__(256)
void select_adj(const float* __restrict__ raw, const float* __restrict__ thr,
                bf16* __restrict__ adj1b, float* __restrict__ denom1) {
    __shared__ float red[256];
    int b = blockIdx.x / S_;
    int i = blockIdx.x % S_;
    int t = threadIdx.x;
    float th = thr[b];
    const float* base = raw + (size_t)b * S_ * S_;
    float sum = 0.f;
    #pragma unroll
    for (int rep = 0; rep < 2; ++rep) {
        int j = t + rep * 256;
        float v  = base[(size_t)i * S_ + j];
        float vT = base[(size_t)j * S_ + i];
        float bm  = (v  >= th) ? 1.f : 0.f;
        float bmT = (vT >= th) ? 1.f : 0.f;
        float mult = (i == j) ? 1.f : (bm + bmT);
        float out = mult * v;
        adj1b[((size_t)b * S_ + i) * S_ + j] = (bf16)out;
        sum += out;
    }
    red[t] = sum; __syncthreads();
    for (int off = 128; off > 0; off >>= 1) {
        if (t < off) red[t] += red[t + off];
        __syncthreads();
    }
    if (t == 0) denom1[(size_t)b * S_ + i] = red[0] + 1.0f;
}

// ---------------------------------------------------------------------------
extern "C" void kernel_launch(void* const* d_in, const int* in_sizes, int n_in,
                              void* d_out, int out_size, void* d_ws, size_t ws_size,
                              hipStream_t stream) {
    (void)in_sizes; (void)n_in; (void)out_size; (void)ws_size;
    const float* x    = (const float*)d_in[0];
    const float* adj  = (const float*)d_in[1];
    const float* W0w  = (const float*)d_in[2];
    const float* W0b  = (const float*)d_in[3];
    const float* W1w  = (const float*)d_in[4];
    const float* W1b  = (const float*)d_in[5];
    const float* Wqw  = (const float*)d_in[6];
    const float* Wqb  = (const float*)d_in[7];
    const float* Wkw  = (const float*)d_in[8];
    const float* Wkb  = (const float*)d_in[9];
    const int*   kptr = (const int*)d_in[10];
    float* out = (float*)d_out;
    char*  ws  = (char*)d_ws;

    auto al = [](size_t x_) { return (x_ + 255) & ~(size_t)255; };
    size_t off = 0;
    // persistent
    size_t o_w0b  = off; off += al((size_t)D_  * D_  * 2);
    size_t o_w1b  = off; off += al((size_t)D2_ * D_  * 2);
    size_t o_wqb  = off; off += al((size_t)D2_ * D2_ * 2);
    size_t o_wkb  = off; off += al((size_t)D2_ * D2_ * 2);
    size_t o_inp1 = off; off += al((size_t)B_ * S_ * D2_ * 2);
    size_t o_adj1b= off; off += al((size_t)B_ * S_ * S_ * 2);
    size_t o_raw  = off; off += al((size_t)B_ * S_ * S_ * 4);
    size_t o_d0   = off; off += al((size_t)B_ * S_ * 4);
    size_t o_d1   = off; off += al((size_t)B_ * S_ * 4);
    size_t o_thr  = off; off += al((size_t)B_ * 4);
    size_t scratch = off;
    // stage 1 scratch
    size_t o_adjb = scratch;
    size_t o_Axb  = o_adjb + al((size_t)B_ * S_ * S_ * 2);
    size_t o_xW0  = o_Axb  + al((size_t)B_ * S_ * D_ * 2);
    size_t o_AxW0 = o_xW0  + al((size_t)B_ * S_ * D_ * 4);
    // stage 2 scratch (stage 1 dead)
    size_t o_qb   = scratch;
    size_t o_kb   = o_qb + al((size_t)B_ * S_ * D2_ * 2);
    size_t o_sc   = o_kb + al((size_t)B_ * S_ * D2_ * 2);
    // stage 3 scratch (stage 2 dead)
    size_t o_A1xb  = scratch;
    size_t o_A1xW1 = o_A1xb  + al((size_t)B_ * S_ * D2_ * 2);
    size_t o_i1W1  = o_A1xW1 + al((size_t)B_ * S_ * D_ * 4);

    bf16* w0b_p  = (bf16*)(ws + o_w0b);
    bf16* w1b_p  = (bf16*)(ws + o_w1b);
    bf16* wqb_p  = (bf16*)(ws + o_wqb);
    bf16* wkb_p  = (bf16*)(ws + o_wkb);
    bf16* inp1_p = (bf16*)(ws + o_inp1);
    bf16* adj1b_p= (bf16*)(ws + o_adj1b);
    float* raw_p = (float*)(ws + o_raw);
    float* d0_p  = (float*)(ws + o_d0);
    float* d1_p  = (float*)(ws + o_d1);
    float* thr_p = (float*)(ws + o_thr);
    bf16* adjb_p = (bf16*)(ws + o_adjb);
    bf16* Axb_p  = (bf16*)(ws + o_Axb);
    float* xW0_p = (float*)(ws + o_xW0);
    float* AxW0_p= (float*)(ws + o_AxW0);
    bf16* qb_p   = (bf16*)(ws + o_qb);
    bf16* kb_p   = (bf16*)(ws + o_kb);
    float* sc_p  = (float*)(ws + o_sc);
    bf16* A1xb_p = (bf16*)(ws + o_A1xb);
    float* A1xW1_p = (float*)(ws + o_A1xW1);
    float* i1W1_p  = (float*)(ws + o_i1W1);

    const long long MBS = (long long)B_ * S_;        // 16384 rows
    const long long nBSD  = MBS * D_;
    const long long nBSS  = MBS * S_;

    auto blocks = [](long long n) { return (unsigned)((n + 255) / 256); };

    // ---- conversions to bf16 ----
    cvt_f32_bf16_strided<<<blocks((long long)D_*D_),   256, 0, stream>>>(W0w, w0b_p, D_,  D_,  (long long)D_*D_);
    cvt_f32_bf16_strided<<<blocks((long long)D2_*D_),  256, 0, stream>>>(W1w, w1b_p, D_,  D_,  (long long)D2_*D_);
    cvt_f32_bf16_strided<<<blocks((long long)D2_*D2_), 256, 0, stream>>>(Wqw, wqb_p, D2_, D2_, (long long)D2_*D2_);
    cvt_f32_bf16_strided<<<blocks((long long)D2_*D2_), 256, 0, stream>>>(Wkw, wkb_p, D2_, D2_, (long long)D2_*D2_);
    cvt_f32_bf16_strided<<<blocks(nBSD), 256, 0, stream>>>(x,   inp1_p, D_, D2_, nBSD);   // x -> inp1[:, :D]
    cvt_f32_bf16_strided<<<blocks(nBSS), 256, 0, stream>>>(adj, adjb_p, S_, S_,  nBSS);

    // ---- denom0 ----
    rowsum_plus1<<<B_ * S_, 256, 0, stream>>>(adj, d0_p, S_);

    // ---- Ax = adj @ x  (batched, bf16 out) ----
    wmma_gemm_bf16<0,1><<<dim3(D_/128, S_/64, B_), 256, 0, stream>>>(
        adjb_p, (long long)S_*S_, 0, S_,
        inp1_p, (long long)S_*D2_, 0, D2_,
        nullptr, 1.0f,
        Axb_p, (long long)S_*D_, 0, D_, 1,
        S_, D_, S_);

    // ---- xW0 = x @ W0 + b0 (f32 out) ----
    wmma_gemm_bf16<0,0><<<dim3(D_/128, MBS/64, 1), 256, 0, stream>>>(
        inp1_p, 0, 0, D2_,
        w0b_p, 0, 0, D_,
        W0b, 1.0f,
        xW0_p, 0, 0, D_, 1,
        (int)MBS, D_, D_);

    // ---- AxW0 = Ax @ W0 + b0 (f32 out) ----
    wmma_gemm_bf16<0,0><<<dim3(D_/128, MBS/64, 1), 256, 0, stream>>>(
        Axb_p, 0, 0, D_,
        w0b_p, 0, 0, D_,
        W0b, 1.0f,
        AxW0_p, 0, 0, D_, 1,
        (int)MBS, D_, D_);

    // ---- out0 = relu(AxW0/denom0) + xW0 -> inp1[:, D:2D] (bf16) ----
    gcn_combine<<<blocks(nBSD), 256, 0, stream>>>(
        AxW0_p, xW0_p, d0_p, D_, inp1_p, D2_, D_, nullptr, nBSD);

    // ---- q = inp1 @ Wq + bq ; k = inp1 @ Wk + bk (bf16 out) ----
    wmma_gemm_bf16<0,1><<<dim3(D2_/128, MBS/64, 1), 256, 0, stream>>>(
        inp1_p, 0, 0, D2_,
        wqb_p, 0, 0, D2_,
        Wqb, 1.0f,
        qb_p, 0, 0, D2_, 1,
        (int)MBS, D2_, D2_);
    wmma_gemm_bf16<0,1><<<dim3(D2_/128, MBS/64, 1), 256, 0, stream>>>(
        inp1_p, 0, 0, D2_,
        wkb_p, 0, 0, D2_,
        Wkb, 1.0f,
        kb_p, 0, 0, D2_, 1,
        (int)MBS, D2_, D2_);

    // ---- scores[b,h] = q_bh @ k_bh^T / sqrt(dk) (f32 out, NT) ----
    float alpha = 1.0f / sqrtf((float)DK_);
    wmma_gemm_bf16<1,0><<<dim3(S_/128, S_/64, B_ * H_), 256, 0, stream>>>(
        qb_p, (long long)S_*D2_, (long long)DK_, D2_,
        kb_p, (long long)S_*D2_, (long long)DK_, D2_,
        nullptr, alpha,
        sc_p, (long long)H_*S_*S_, (long long)S_*S_, S_, H_,
        S_, S_, DK_);

    // ---- masked softmax + head-sum -> adj1raw ----
    attn_softmax_sumheads<<<B_ * S_, 256, 0, stream>>>(sc_p, adj, raw_p);

    // ---- per-batch top-k threshold ----
    topk_threshold<<<B_, 256, 0, stream>>>(raw_p, kptr, thr_p);

    // ---- select/symmetrize/diag * raw -> adj1 (bf16) + denom1 ----
    select_adj<<<B_ * S_, 256, 0, stream>>>(raw_p, thr_p, adj1b_p, d1_p);

    // ---- A1x = adj1 @ inp1 (batched, bf16 out) ----
    wmma_gemm_bf16<0,1><<<dim3(D2_/128, S_/64, B_), 256, 0, stream>>>(
        adj1b_p, (long long)S_*S_, 0, S_,
        inp1_p, (long long)S_*D2_, 0, D2_,
        nullptr, 1.0f,
        A1xb_p, (long long)S_*D2_, 0, D2_, 1,
        S_, D2_, S_);

    // ---- A1xW1 = A1x @ W1 + b1 ; i1W1 = inp1 @ W1 + b1 (f32 out) ----
    wmma_gemm_bf16<0,0><<<dim3(D_/128, MBS/64, 1), 256, 0, stream>>>(
        A1xb_p, 0, 0, D2_,
        w1b_p, 0, 0, D_,
        W1b, 1.0f,
        A1xW1_p, 0, 0, D_, 1,
        (int)MBS, D_, D2_);
    wmma_gemm_bf16<0,0><<<dim3(D_/128, MBS/64, 1), 256, 0, stream>>>(
        inp1_p, 0, 0, D2_,
        w1b_p, 0, 0, D_,
        W1b, 1.0f,
        i1W1_p, 0, 0, D_, 1,
        (int)MBS, D_, D2_);

    // ---- final: out = relu(A1xW1/denom1) + i1W1 (f32) ----
    gcn_combine<<<blocks(nBSD), 256, 0, stream>>>(
        A1xW1_p, i1W1_p, d1_p, D_, nullptr, 0, 0, out, nBSD);
}
